// ScaledDotProductAttention_214748365181
// MI455X (gfx1250) — matrix operand order repaired and checked by reading, not compiled, not run on hardware
//
#include <hip/hip_runtime.h>
#include <cmath>

typedef __attribute__((ext_vector_type(2)))  float    v2f;
typedef __attribute__((ext_vector_type(8)))  float    v8f;
typedef __attribute__((ext_vector_type(16))) _Float16 v16h;

#define B_    2
#define H_    16
#define S_    2048
#define D_    64
#define QTILE 16
#define NWAVES 4
#define NTHREADS 128
#define NEGV  (-1.0e9f)
#define SCALE (0.125f)   /* 1/sqrt(64) */

__launch_bounds__(NTHREADS, 1)
__global__ void sdpa_fp32_wmma_kernel(const float* __restrict__ Q,
                                      const float* __restrict__ K,
                                      const float* __restrict__ V,
                                      const unsigned char* __restrict__ Mask,
                                      float* __restrict__ Out,
                                      float* __restrict__ Wout) {
  // 128 KB static LDS score tile (gfx1250 allows 320 KB/workgroup)
  __shared__ float sc[QTILE * S_];
  __shared__ float red[8 * 16];
  __shared__ float rowmax[16];
  __shared__ float rinv[16];

  const int tid  = threadIdx.x;
  const int wave = tid >> 5;
  const int lane = tid & 31;             // wave32
  const int ll   = lane & 15;
  const int hi   = lane >> 4;            // 0 or 1 (upper half-wave)

  const int blk = blockIdx.x;
  const int bh  = blk >> 7;              // / (S_/QTILE = 128)
  const int qt  = blk & 127;
  const int b   = bh >> 4;               // / H_
  const int q0  = qt * QTILE;

  const float* qbase = Q + (size_t)bh * S_ * D_;
  const float* kbase = K + (size_t)bh * S_ * D_;
  const float* vbase = V + (size_t)bh * S_ * D_;
  const unsigned char* mbase = Mask + (size_t)b * S_ * S_ + (size_t)q0 * S_;

  // ---------------- Phase 1: S = (Q K^T) * scale, masked -> LDS ----------------
  // A (16x4 f32): lanes 0-15 = M rows, VGPR0/1 = K {0,1}; lanes 16-31 = K {2,3}.
  v2f aq[16];
  {
    const float* qrow = qbase + (size_t)(q0 + ll) * D_ + 2 * hi;
    #pragma unroll
    for (int t = 0; t < 16; ++t) {
      aq[t].x = qrow[4 * t + 0];
      aq[t].y = qrow[4 * t + 1];
    }
  }
  // Two N-tiles per iteration -> two independent WMMA accumulation chains (ILP).
  for (int nt = 2 * wave; nt < S_ / 16; nt += 2 * NWAVES) {
    v8f c0 = {};
    v8f c1 = {};
    const float* krow0 = kbase + (size_t)(nt * 16 + ll) * D_ + 2 * hi;
    const float* krow1 = krow0 + 16 * D_;
    #pragma unroll
    for (int t = 0; t < 16; ++t) {
      v2f bf0, bf1;
      bf0.x = krow0[4 * t + 0];
      bf0.y = krow0[4 * t + 1];
      bf1.x = krow1[4 * t + 0];
      bf1.y = krow1[4 * t + 1];
      c0 = __builtin_amdgcn_wmma_f32_16x16x4_f32(false, aq[t], false, bf0,
                                                 (short)0, c0, false, false);
      c1 = __builtin_amdgcn_wmma_f32_16x16x4_f32(false, aq[t], false, bf1,
                                                 (short)0, c1, false, false);
    }
    // C/D layout: lane ll = column N, VGPR i -> row M = i + 8*hi
    #pragma unroll
    for (int i = 0; i < 8; ++i) {
      const int m  = i + 8 * hi;
      const int n0 = nt * 16 + ll;
      float s0 = c0[i] * SCALE;
      float s1 = c1[i] * SCALE;
      if (mbase[(size_t)m * S_ + n0])      s0 = NEGV;  // mask==True -> -1e9
      if (mbase[(size_t)m * S_ + n0 + 16]) s1 = NEGV;
      sc[m * S_ + n0]      = s0;
      sc[m * S_ + n0 + 16] = s1;
    }
  }
  __syncthreads();

  // ---------------- Phase 2: softmax (LDS keeps unnormalized exp) ----------------
  const int r  = tid & 15;     // row handled by this thread
  const int ch = tid >> 4;     // 0..7 column chunk (256 cols = 64 float4 each)
  {
    const float4* rp4 = (const float4*)(sc + (size_t)r * S_) + ch * 64;
    float4 mx = rp4[0];
    #pragma unroll 4
    for (int n = 1; n < 64; ++n) {
      float4 t = rp4[n];
      mx.x = fmaxf(mx.x, t.x); mx.y = fmaxf(mx.y, t.y);
      mx.z = fmaxf(mx.z, t.z); mx.w = fmaxf(mx.w, t.w);
    }
    red[ch * 16 + r] = fmaxf(fmaxf(mx.x, mx.y), fmaxf(mx.z, mx.w));
  }
  __syncthreads();
  if (tid < 16) {
    float mm = red[tid];
    #pragma unroll
    for (int j = 1; j < 8; ++j) mm = fmaxf(mm, red[j * 16 + tid]);
    rowmax[tid] = mm;
  }
  __syncthreads();
  {
    const float rm = rowmax[r];
    float4* rp4 = (float4*)(sc + (size_t)r * S_) + ch * 64;
    float4 acc = {0.f, 0.f, 0.f, 0.f};
    #pragma unroll 4
    for (int n = 0; n < 64; ++n) {
      float4 t = rp4[n];
      t.x = __expf(t.x - rm); t.y = __expf(t.y - rm);
      t.z = __expf(t.z - rm); t.w = __expf(t.w - rm);
      rp4[n] = t;
      acc.x += t.x; acc.y += t.y; acc.z += t.z; acc.w += t.w;
    }
    red[ch * 16 + r] = (acc.x + acc.y) + (acc.z + acc.w);
  }
  __syncthreads();
  if (tid < 16) {
    float ss = red[tid];
    #pragma unroll
    for (int j = 1; j < 8; ++j) ss += red[j * 16 + tid];
    rinv[tid] = 1.0f / ss;
  }
  __syncthreads();

  // ---------------- Phase 3: write attn_weights = exp * inv (coalesced float4) ----
  {
    float* wrow = Wout + ((size_t)bh * S_ + q0) * S_;
    for (int idx = tid; idx < QTILE * (S_ / 4); idx += NTHREADS) {
      const int mrow = idx >> 9;         // / 512 float4s per row
      const int n4   = idx & 511;
      float4 w4 = ((const float4*)(sc + (size_t)mrow * S_))[n4];
      const float iv = rinv[mrow];
      w4.x *= iv; w4.y *= iv; w4.z *= iv; w4.w *= iv;
      ((float4*)(wrow + (size_t)mrow * S_))[n4] = w4;
    }
  }

  // ---------------- Phase 4: Out = P V via v_wmma_f32_16x16x32_f16 ----------------
  // wave w owns output columns [16*w, 16*w+16); two interleaved accumulator
  // chains over the K loop (kb and kb+32) to break the WMMA RAW chain.
  {
    const int dbase = wave * 16;
    v8f c0 = {};
    v8f c1 = {};
    const float4* arow4 = (const float4*)(sc + (size_t)ll * S_);
    for (int kb = 0; kb < S_; kb += 64) {
      v16h a0h, a1h, bv0, bv1;
      #pragma unroll
      for (int half = 0; half < 2; ++half) {
        const int kbb = kb + 32 * half;
        // A 16x32 f16 layout: halves j<8 -> K = kbb + 8*hi + j ;
        //                     halves j>=8 -> K = kbb + 16 + 8*hi + (j-8)
        const int i0 = (kbb + 8 * hi) >> 2;
        float4 a0 = arow4[i0];
        float4 a1 = arow4[i0 + 1];
        float4 a2 = arow4[i0 + 4];           // +16 floats
        float4 a3 = arow4[i0 + 5];
        v16h a;
        a[0]  = (_Float16)a0.x; a[1]  = (_Float16)a0.y;
        a[2]  = (_Float16)a0.z; a[3]  = (_Float16)a0.w;
        a[4]  = (_Float16)a1.x; a[5]  = (_Float16)a1.y;
        a[6]  = (_Float16)a1.z; a[7]  = (_Float16)a1.w;
        a[8]  = (_Float16)a2.x; a[9]  = (_Float16)a2.y;
        a[10] = (_Float16)a2.z; a[11] = (_Float16)a2.w;
        a[12] = (_Float16)a3.x; a[13] = (_Float16)a3.y;
        a[14] = (_Float16)a3.z; a[15] = (_Float16)a3.w;
        // B 32x16 f16 layout: lane ll = column N, half j -> K = kbb + 8*hi + (j&7) + 16*(j>>3)
        v16h bv;
        #pragma unroll
        for (int j = 0; j < 16; ++j) {
          const int kr = kbb + (j & 7) + 8 * hi + ((j >> 3) << 4);
          bv[j] = (_Float16)vbase[(size_t)kr * D_ + dbase + ll];
        }
        if (half == 0) { a0h = a; bv0 = bv; } else { a1h = a; bv1 = bv; }
      }
      c0 = __builtin_amdgcn_wmma_f32_16x16x32_f16(false, a0h, false, bv0,
                                                  (short)0, c0, false, false);
      c1 = __builtin_amdgcn_wmma_f32_16x16x32_f16(false, a1h, false, bv1,
                                                  (short)0, c1, false, false);
    }
    #pragma unroll
    for (int i = 0; i < 8; ++i) {
      const int m = i + 8 * hi;
      Out[((size_t)bh * S_ + q0 + m) * D_ + dbase + ll] = (c0[i] + c1[i]) * rinv[m];
    }
  }
}

extern "C" void kernel_launch(void* const* d_in, const int* in_sizes, int n_in,
                              void* d_out, int out_size, void* d_ws, size_t ws_size,
                              hipStream_t stream) {
  (void)in_sizes; (void)n_in; (void)out_size; (void)d_ws; (void)ws_size;
  const float* q = (const float*)d_in[0];
  const float* k = (const float*)d_in[1];
  const float* v = (const float*)d_in[2];
  const unsigned char* m = (const unsigned char*)d_in[3];   // jnp.bool_ -> 1 byte

  float* out  = (float*)d_out;                               // [B,H,S,D] first
  float* wout = out + (size_t)B_ * H_ * S_ * D_;             // then [B,H,S,S]

  dim3 grid(B_ * H_ * (S_ / QTILE));   // 4096 blocks
  dim3 block(NTHREADS);                // 4 waves (wave32)
  sdpa_fp32_wmma_kernel<<<grid, block, 0, stream>>>(q, k, v, m, out, wout);
}
